// MultiHeadSelfAttention_57380763074931
// MI455X (gfx1250) — compile-verified, matrix operand order checked by reading, not run on hardware
//
#include <hip/hip_runtime.h>

#define SEQ 4096
#define DM  1024
#define NH  16
#define DH  64

typedef __attribute__((ext_vector_type(16))) __bf16 v16bf;
typedef __attribute__((ext_vector_type(8)))  float  v8f;

union FragAB {
  v16bf v;
  uint4 u[2];
  unsigned short s[16];
};

__device__ __forceinline__ unsigned short f2bf(float f) {
  unsigned int u = __float_as_uint(f);
  u += 0x7FFFu + ((u >> 16) & 1u);     // round-to-nearest-even
  return (unsigned short)(u >> 16);
}

__device__ __forceinline__ v8f wmma_bf16(const FragAB& a, const FragAB& b, v8f c) {
  return __builtin_amdgcn_wmma_f32_16x16x32_bf16(
      /*neg_a=*/false, a.v, /*neg_b=*/false, b.v,
      /*c_mod=*/(short)0, c, /*reuse_a=*/false, /*reuse_b=*/false);
}

__device__ __forceinline__ void load_frag(FragAB& f, const unsigned short* p) {
  f.u[0] = *(const uint4*)p;        // K run 0: 8 bf16
  f.u[1] = *(const uint4*)(p + 16); // K run 1: 8 bf16
}

__global__ void f32_to_bf16_kernel(const float* __restrict__ src,
                                   unsigned short* __restrict__ dst, int n) {
  int i = blockIdx.x * blockDim.x + threadIdx.x;
  if (i < n) dst[i] = f2bf(src[i]);
}

// ---------------- GEMM: C(MxN) = A(MxK) @ B(NxK)^T + bias ----------------
// A,B bf16 row-major (contraction contiguous per lane).
// MODE 0: store bf16 row-major (ld=N)
// MODE 1: store bf16 transposed Ct[n*M + m]  (for V^T)
// MODE 2: store f32 row-major (ld=N)
// One wave computes a 64x32 tile (4x2 WMMA tiles), double-buffered over K.

__device__ __forceinline__ void load_tileAB(FragAB (&a)[4], FragAB (&b)[2],
                                            const unsigned short* __restrict__ A,
                                            const unsigned short* __restrict__ B,
                                            int mb, int nb, int K, int k0,
                                            int lo, int kh) {
#pragma unroll
  for (int i = 0; i < 4; ++i)
    load_frag(a[i], A + (size_t)(mb + 16 * i + lo) * K + k0 + kh * 8);
#pragma unroll
  for (int j = 0; j < 2; ++j)
    load_frag(b[j], B + (size_t)(nb + 16 * j + lo) * K + k0 + kh * 8);
}

__device__ __forceinline__ void mac_tile(v8f (&acc)[4][2],
                                         const FragAB (&a)[4],
                                         const FragAB (&b)[2]) {
#pragma unroll
  for (int i = 0; i < 4; ++i)
#pragma unroll
    for (int j = 0; j < 2; ++j)
      acc[i][j] = wmma_bf16(a[i], b[j], acc[i][j]);
}

template <int MODE>
__global__ __launch_bounds__(32, 1)
void gemm_bf16_wmma(const unsigned short* __restrict__ A,
                    const unsigned short* __restrict__ B,
                    const float* __restrict__ bias,
                    void* __restrict__ Cout,
                    int M, int N, int K) {
  const int lane = threadIdx.x;
  const int lo = lane & 15;
  const int kh = lane >> 4;          // which K-half this lane holds
  const int nb = blockIdx.x * 32;
  const int mb = blockIdx.y * 64;

  v8f zero = {};
  v8f acc[4][2];
#pragma unroll
  for (int i = 0; i < 4; ++i)
#pragma unroll
    for (int j = 0; j < 2; ++j) acc[i][j] = zero;

  FragAB a0[4], b0[2], a1[4], b1[2];

  // software pipeline: K assumed multiple of 64 (K = 1024 here)
  load_tileAB(a0, b0, A, B, mb, nb, K, 0, lo, kh);
  for (int k0 = 0; k0 < K; k0 += 64) {
    load_tileAB(a1, b1, A, B, mb, nb, K, k0 + 32, lo, kh);
    mac_tile(acc, a0, b0);
    if (k0 + 64 < K) load_tileAB(a0, b0, A, B, mb, nb, K, k0 + 64, lo, kh);
    mac_tile(acc, a1, b1);
  }

#pragma unroll
  for (int j = 0; j < 2; ++j) {
    const float bn = bias[nb + 16 * j + lo];
#pragma unroll
    for (int i = 0; i < 4; ++i) {
      if (MODE == 1) {
        unsigned short tmp[8];
#pragma unroll
        for (int r = 0; r < 8; ++r) tmp[r] = f2bf(acc[i][j][r] + bn);
        const int n  = nb + 16 * j + lo;
        const int m0 = mb + 16 * i + 8 * kh;   // 8 consecutive rows per lane
        *(uint4*)((unsigned short*)Cout + (size_t)n * M + m0) = *(const uint4*)tmp;
      } else {
#pragma unroll
        for (int r = 0; r < 8; ++r) {
          const int m = mb + 16 * i + 8 * kh + r;
          const int n = nb + 16 * j + lo;
          const float val = acc[i][j][r] + bn;
          if (MODE == 0)
            ((unsigned short*)Cout)[(size_t)m * N + n] = f2bf(val);
          else
            ((float*)Cout)[(size_t)m * N + n] = val;
        }
      }
    }
  }
}

// ---------------- Flash attention, causal ----------------
// One wave per (16-query tile, head).
// Q,K: (SEQ, DM) bf16 row-major.  Vt: (DM, SEQ) bf16 (V transposed).  O: (SEQ, DM) bf16.
__global__ __launch_bounds__(32, 1)
void flash_attn_kernel(const unsigned short* __restrict__ Q,
                       const unsigned short* __restrict__ Km,
                       const unsigned short* __restrict__ Vt,
                       unsigned short* __restrict__ O) {
  __shared__ float pl[16][32];
  const int lane = threadIdx.x;
  const int lo = lane & 15;
  const int kh = lane >> 4;
  const int qb = blockIdx.x * 16;
  const int h  = blockIdx.y;
  const int hd = h * DH;

  // Q fragments: 16 rows x 64 (two K=32 steps)
  FragAB aq[2];
#pragma unroll
  for (int i = 0; i < 2; ++i)
    load_frag(aq[i], Q + (size_t)(qb + lo) * DM + hd + i * 32 + kh * 8);

  v8f zero = {};
  v8f oacc[4] = {zero, zero, zero, zero};   // 16 x 64 output accumulator
  float rm[8], rl[8];                        // per-row running max / sum (rows r+8*kh)
#pragma unroll
  for (int r = 0; r < 8; ++r) { rm[r] = -1e30f; rl[r] = 0.0f; }

  const int kend = qb + 16;                  // causal: keys <= last query row
  for (int kb = 0; kb < kend; kb += 32) {
    // ---- S = Q K^T for a 16x32 tile of scores ----
    // K-frags loaded per half; their live range ends at the WMMA that uses them.
    float s[2][8];
#pragma unroll
    for (int c = 0; c < 2; ++c) {
      FragAB kf0, kf1;
      load_frag(kf0, Km + (size_t)(kb + c * 16 + lo) * DM + hd + 0  + kh * 8);
      load_frag(kf1, Km + (size_t)(kb + c * 16 + lo) * DM + hd + 32 + kh * 8);
      v8f sacc = zero;
      sacc = wmma_bf16(aq[0], kf0, sacc);
      sacc = wmma_bf16(aq[1], kf1, sacc);
      const int col = kb + c * 16 + lo;
#pragma unroll
      for (int r = 0; r < 8; ++r) {
        const int row = qb + 8 * kh + r;
        const float v = sacc[r] * 0.125f;          // 1/sqrt(64)
        s[c][r] = (col > row) ? -1e30f : v;        // causal mask (select, EXEC intact)
      }
    }

    // ---- issue V loads now; the softmax below hides their latency ----
    FragAB vf[4];
#pragma unroll
    for (int dc = 0; dc < 4; ++dc)
      load_frag(vf[dc], Vt + (size_t)(hd + dc * 16 + lo) * SEQ + kb + kh * 8);

    // ---- online softmax update (row lives in one 16-lane half) ----
#pragma unroll
    for (int r = 0; r < 8; ++r) {
      float tmax = fmaxf(s[0][r], s[1][r]);
#pragma unroll
      for (int off = 1; off < 16; off <<= 1)
        tmax = fmaxf(tmax, __shfl_xor(tmax, off, 32));
      const float mn = fmaxf(rm[r], tmax);
      const float alpha = __expf(rm[r] - mn);
      rm[r] = mn;
      const float p0 = __expf(s[0][r] - mn);
      const float p1 = __expf(s[1][r] - mn);
      s[0][r] = p0; s[1][r] = p1;
      float ps = p0 + p1;
#pragma unroll
      for (int off = 1; off < 16; off <<= 1)
        ps += __shfl_xor(ps, off, 32);
      rl[r] = rl[r] * alpha + ps;
#pragma unroll
      for (int dc = 0; dc < 4; ++dc) oacc[dc][r] *= alpha;
    }

    // ---- P tile (C-layout) -> A-fragment layout via LDS ----
    __syncthreads();
#pragma unroll
    for (int c = 0; c < 2; ++c)
#pragma unroll
      for (int r = 0; r < 8; ++r)
        pl[8 * kh + r][c * 16 + lo] = s[c][r];
    __syncthreads();
    FragAB ap;
#pragma unroll
    for (int e = 0; e < 8; ++e) {
      ap.s[e]     = f2bf(pl[lo][8 * kh + e]);        // K run 0
      ap.s[8 + e] = f2bf(pl[lo][16 + 8 * kh + e]);   // K run 1
    }

    // ---- O += P @ V  (V-frags issued before the softmax; wait is cheap) ----
#pragma unroll
    for (int dc = 0; dc < 4; ++dc)
      oacc[dc] = wmma_bf16(ap, vf[dc], oacc[dc]);
  }

  // ---- normalize and store O (bf16 row-major) ----
#pragma unroll
  for (int r = 0; r < 8; ++r) {
    const float inv = 1.0f / rl[r];
    const int row = qb + 8 * kh + r;
#pragma unroll
    for (int dc = 0; dc < 4; ++dc)
      O[(size_t)row * DM + hd + dc * 16 + lo] = f2bf(oacc[dc][r] * inv);
  }
}

extern "C" void kernel_launch(void* const* d_in, const int* in_sizes, int n_in,
                              void* d_out, int out_size, void* d_ws, size_t ws_size,
                              hipStream_t stream) {
  const float* x  = (const float*)d_in[0];
  // d_in[1] = causal mask (bool) — computed analytically, unused
  const float* Wq = (const float*)d_in[2];
  const float* bq = (const float*)d_in[3];
  const float* Wk = (const float*)d_in[4];
  const float* bk = (const float*)d_in[5];
  const float* Wv = (const float*)d_in[6];
  const float* bv = (const float*)d_in[7];
  const float* Wo = (const float*)d_in[8];
  const float* bo = (const float*)d_in[9];
  float* out = (float*)d_out;

  // workspace carve-up (bf16 buffers; 48 MB total, fits L2)
  unsigned short* xb  = (unsigned short*)d_ws;
  unsigned short* Wqb = xb  + (size_t)SEQ * DM;
  unsigned short* Wkb = Wqb + (size_t)DM * DM;
  unsigned short* Wvb = Wkb + (size_t)DM * DM;
  unsigned short* Wob = Wvb + (size_t)DM * DM;
  unsigned short* Qb  = Wob + (size_t)DM * DM;
  unsigned short* Kb  = Qb  + (size_t)SEQ * DM;
  unsigned short* Vt  = Kb  + (size_t)SEQ * DM;
  unsigned short* Ob  = Vt  + (size_t)SEQ * DM;

  const int nx = SEQ * DM;
  const int nw = DM * DM;
  f32_to_bf16_kernel<<<(nx + 255) / 256, 256, 0, stream>>>(x, xb, nx);
  f32_to_bf16_kernel<<<(nw + 255) / 256, 256, 0, stream>>>(Wq, Wqb, nw);
  f32_to_bf16_kernel<<<(nw + 255) / 256, 256, 0, stream>>>(Wk, Wkb, nw);
  f32_to_bf16_kernel<<<(nw + 255) / 256, 256, 0, stream>>>(Wv, Wvb, nw);
  f32_to_bf16_kernel<<<(nw + 255) / 256, 256, 0, stream>>>(Wo, Wob, nw);

  dim3 gg(DM / 32, SEQ / 64);
  gemm_bf16_wmma<0><<<gg, 32, 0, stream>>>(xb, Wqb, bq, Qb, SEQ, DM, DM);
  gemm_bf16_wmma<0><<<gg, 32, 0, stream>>>(xb, Wkb, bk, Kb, SEQ, DM, DM);
  gemm_bf16_wmma<1><<<gg, 32, 0, stream>>>(xb, Wvb, bv, Vt, SEQ, DM, DM);

  flash_attn_kernel<<<dim3(SEQ / 16, NH), 32, 0, stream>>>(Qb, Kb, Vt, Ob);

  gemm_bf16_wmma<2><<<gg, 32, 0, stream>>>(Ob, Wob, bo, out, SEQ, DM, DM);
}
